// LSTMCell_29686813950554
// MI455X (gfx1250) — matrix operand order repaired and checked by reading, not compiled, not run on hardware
//
#include <hip/hip_runtime.h>
#include <math.h>

// ---------------------------------------------------------------------------
// LSTM cell for gfx1250 (CDNA5, wave32, v_wmma_f32_16x16x32_f16).
//
// Pass 1: convert input/hidden fp32->f16 into ws; convert+transpose the 8
//         weight matrices fp32[k][n] -> f16[n][k] into ws (one-time cost).
// Pass 2: fused dual-GEMM (x-path + h-path) over all 4 gates with a 128x64
//         workgroup tile, double-buffered LDS, software-pipelined staging,
//         and fused sigmoid/tanh/cell-update epilogue.
// ws layout (f16): [0, 4M)   input
//                  [4M, 8M)  hidden
//                  [8M, 16M) Wt[phase][gate f,i,n,o][n][k]  (8 x 1Mx f16)
// ---------------------------------------------------------------------------

typedef __attribute__((ext_vector_type(16))) _Float16 v16h;
typedef __attribute__((ext_vector_type(8)))  _Float16 v8h;
typedef __attribute__((ext_vector_type(4)))  _Float16 v4h;
typedef __attribute__((ext_vector_type(8)))  float    v8f;

#define BATCH_SZ 4096
#define HID      1024
#define BM       128
#define BN       64
#define BK       32
#define LDA      40            // BK + 8 halves pad (80B rows, 16B aligned)
#define LDB      40
#define ACT_SZ   (BATCH_SZ * HID)       // 4M halves per activation matrix
#define MAT_SZ   (HID * HID)            // 1M halves per weight matrix
#define NSTEP    (2 * (HID / BK))       // 64 k-steps total (x-path + h-path)

__device__ __forceinline__ float sigmoidf_(float x) {
    return 1.0f / (1.0f + __expf(-x));
}

__device__ __forceinline__ v16h cat16(v8h lo, v8h hi) {
    return __builtin_shufflevector(lo, hi, 0, 1, 2, 3, 4, 5, 6, 7,
                                           8, 9, 10, 11, 12, 13, 14, 15);
}

// ---- pass 1a: fp32 -> f16 straight conversion (input / hidden) -------------
__global__ __launch_bounds__(256)
void cvt_act(const float* __restrict__ src, _Float16* __restrict__ dst) {
    const int i = (blockIdx.x * 256 + threadIdx.x) * 4;
    float4 v = *(const float4*)(src + i);
    v4h h = {(_Float16)v.x, (_Float16)v.y, (_Float16)v.z, (_Float16)v.w};
    *(v4h*)(dst + i) = h;
}

// ---- pass 1b: fp32 [k][n] -> f16 [n][k] transpose for the 8 weights --------
__global__ __launch_bounds__(256)
void cvt_wt(const float* __restrict__ w0, const float* __restrict__ w1,
            const float* __restrict__ w2, const float* __restrict__ w3,
            const float* __restrict__ w4, const float* __restrict__ w5,
            const float* __restrict__ w6, const float* __restrict__ w7,
            _Float16* __restrict__ dst) {
    __shared__ __align__(16) _Float16 t[64][72];   // [k][n], padded

    const float* src;
    switch (blockIdx.z) {
        case 0: src = w0; break; case 1: src = w1; break;
        case 2: src = w2; break; case 3: src = w3; break;
        case 4: src = w4; break; case 5: src = w5; break;
        case 6: src = w6; break; default: src = w7; break;
    }
    _Float16* dstm = dst + (size_t)blockIdx.z * MAT_SZ;

    const int tid = threadIdx.x;
    const int k0  = blockIdx.y * 64;
    const int n0  = blockIdx.x * 64;

    // coalesced read along n, convert, stash in LDS [k][n]
    const int kr  = tid >> 2;
    const int ncg = (tid & 3) * 16;
#pragma unroll
    for (int j = 0; j < 4; ++j) {
        float4 v = *(const float4*)(src + (size_t)(k0 + kr) * HID + n0 + ncg + j * 4);
        v4h h = {(_Float16)v.x, (_Float16)v.y, (_Float16)v.z, (_Float16)v.w};
        *(v4h*)(&t[kr][ncg + j * 4]) = h;
    }
    __syncthreads();

    // transposed write: coalesced along k in the destination [n][k]
    const int nr  = tid >> 2;
    const int kcg = (tid & 3) * 16;
    v8h lo, hi;
#pragma unroll
    for (int e = 0; e < 8; ++e) lo[e] = t[kcg + e][nr];
#pragma unroll
    for (int e = 0; e < 8; ++e) hi[e] = t[kcg + 8 + e][nr];
    _Float16* op = dstm + (size_t)(n0 + nr) * HID + k0 + kcg;
    *(v8h*)(op)     = lo;
    *(v8h*)(op + 8) = hi;
}

// ---- pass 2: fused dual-GEMM LSTM cell -------------------------------------
__global__ __launch_bounds__(256)
void lstm_cell_wmma(const _Float16* __restrict__ hact,  // [2][4096][1024] f16
                    const _Float16* __restrict__ hwt,   // [8][1024][1024] f16 (n-major)
                    const float* __restrict__ cell,
                    const float* __restrict__ b_f, const float* __restrict__ b_i,
                    const float* __restrict__ b_n, const float* __restrict__ b_o,
                    float* __restrict__ out)
{
    __shared__ __align__(16) _Float16 sA[2][BM][LDA];     // [buf][m][k]
    __shared__ __align__(16) _Float16 sB[2][4][BN][LDB];  // [buf][gate][n][k]

    const int tid  = threadIdx.x;
    const int lane = tid & 31;
    const int wid  = tid >> 5;
    const int mwave = (wid >> 1) * 32;
    const int nwave = (wid & 1) * 32;
    const int khalf = lane >> 4;
    const int lm    = lane & 15;

    const int n0 = blockIdx.x * BN;
    const int m0 = blockIdx.y * BM;

    v8f acc[4][2][2];
#pragma unroll
    for (int g = 0; g < 4; ++g)
#pragma unroll
        for (int ms = 0; ms < 2; ++ms)
#pragma unroll
            for (int ns = 0; ns < 2; ++ns)
                acc[g][ms][ns] = (v8f){0.f, 0.f, 0.f, 0.f, 0.f, 0.f, 0.f, 0.f};

    // staging thread maps (all 16B-aligned accesses)
    const int arow = tid >> 1;                 // 0..127
    const int acol = (tid & 1) * 16;           // 0 or 16 halves
    const int brow = tid >> 2;                 // 0..63  (n)
    const int bcol = (tid & 3) * 8;            // k chunk, 8 halves
    const size_t aoff = (size_t)(m0 + arow) * HID + acol;
    const size_t boff = (size_t)(n0 + brow) * HID + bcol;

    v8h aS0, aS1, bS[4];

    // prologue: stage step 0 into buffer 0
    {
        const _Float16* ap = hact + aoff;            // phase 0, kk = 0
        aS0 = *(const v8h*)(ap);
        aS1 = *(const v8h*)(ap + 8);
#pragma unroll
        for (int g = 0; g < 4; ++g)
            bS[g] = *(const v8h*)(hwt + (size_t)g * MAT_SZ + boff);
        *(v8h*)(&sA[0][arow][acol])     = aS0;
        *(v8h*)(&sA[0][arow][acol + 8]) = aS1;
#pragma unroll
        for (int g = 0; g < 4; ++g)
            *(v8h*)(&sB[0][g][brow][bcol]) = bS[g];
    }
    __syncthreads();

#pragma unroll 1
    for (int s = 0; s < NSTEP; ++s) {
        const int cur = s & 1;
        const int nxt = cur ^ 1;

        // -- issue next step's global loads (latency hidden behind WMMAs)
        if (s + 1 < NSTEP) {
            const int phase = (s + 1) >> 5;
            const int kk    = ((s + 1) & 31) * BK;
            const _Float16* ap = hact + (size_t)phase * ACT_SZ + aoff + kk;
            aS0 = *(const v8h*)(ap);
            aS1 = *(const v8h*)(ap + 8);
#pragma unroll
            for (int g = 0; g < 4; ++g)
                bS[g] = *(const v8h*)(hwt + (size_t)(phase * 4 + g) * MAT_SZ +
                                      boff + kk);
        }

        // -- compute on current buffer: 16 WMMAs
        v16h afrag[2];
#pragma unroll
        for (int ms = 0; ms < 2; ++ms) {
            const int row = mwave + ms * 16 + lm;
            const int k0  = khalf * 8;
            v8h lo = *(const v8h*)(&sA[cur][row][k0]);
            v8h hi = *(const v8h*)(&sA[cur][row][k0 + 16]);
            afrag[ms] = cat16(lo, hi);
        }
#pragma unroll
        for (int g = 0; g < 4; ++g) {
#pragma unroll
            for (int ns = 0; ns < 2; ++ns) {
                const int col = nwave + ns * 16 + lm;
                const int klo = khalf * 8;
                v8h blo = *(const v8h*)(&sB[cur][g][col][klo]);
                v8h bhi = *(const v8h*)(&sB[cur][g][col][klo + 16]);
                v16h bfrag = cat16(blo, bhi);
#pragma unroll
                for (int ms = 0; ms < 2; ++ms) {
                    acc[g][ms][ns] = __builtin_amdgcn_wmma_f32_16x16x32_f16(
                        false, afrag[ms], false, bfrag,
                        (short)0, acc[g][ms][ns], false, false);
                }
            }
        }

        // -- commit staged registers into the other buffer
        if (s + 1 < NSTEP) {
            *(v8h*)(&sA[nxt][arow][acol])     = aS0;
            *(v8h*)(&sA[nxt][arow][acol + 8]) = aS1;
#pragma unroll
            for (int g = 0; g < 4; ++g)
                *(v8h*)(&sB[nxt][g][brow][bcol]) = bS[g];
        }
        __syncthreads();
    }

    // ---- epilogue
    const size_t BH = (size_t)BATCH_SZ * HID;
#pragma unroll
    for (int ms = 0; ms < 2; ++ms) {
#pragma unroll
        for (int ns = 0; ns < 2; ++ns) {
            const int colg = n0 + nwave + ns * 16 + lm;
            const float bf = b_f[colg];
            const float bi = b_i[colg];
            const float bnn = b_n[colg];
            const float bo = b_o[colg];
#pragma unroll
            for (int r = 0; r < 8; ++r) {
                const int rowg = m0 + mwave + ms * 16 + r + khalf * 8;
                const size_t idx = (size_t)rowg * HID + colg;
                const float c  = cell[idx];
                const float f  = sigmoidf_(acc[0][ms][ns][r] + bf);
                const float ig = sigmoidf_(acc[1][ms][ns][r] + bi);
                const float ng = tanhf(acc[2][ms][ns][r] + bnn);
                const float og = sigmoidf_(acc[3][ms][ns][r] + bo);
                const float cn = f * c + ig * ng;
                const float hn = og * tanhf(cn);
                out[idx]      = hn;   // new_hidden
                out[BH + idx] = cn;   // new_cell
            }
        }
    }
}

extern "C" void kernel_launch(void* const* d_in, const int* in_sizes, int n_in,
                              void* d_out, int out_size, void* d_ws, size_t ws_size,
                              hipStream_t stream)
{
    (void)in_sizes; (void)n_in; (void)out_size; (void)ws_size;
    // setup_inputs order:
    // 0 input, 1 hidden, 2 cell, 3 W_ii, 4 W_hi, 5 b_i, 6 W_if, 7 W_hf, 8 b_f,
    // 9 W_in, 10 W_hn, 11 b_n, 12 W_io, 13 W_ho, 14 b_o
    const float* input  = (const float*)d_in[0];
    const float* hidden = (const float*)d_in[1];
    const float* cell   = (const float*)d_in[2];
    const float* W_ii   = (const float*)d_in[3];
    const float* W_hi   = (const float*)d_in[4];
    const float* b_i    = (const float*)d_in[5];
    const float* W_if   = (const float*)d_in[6];
    const float* W_hf   = (const float*)d_in[7];
    const float* b_f    = (const float*)d_in[8];
    const float* W_in   = (const float*)d_in[9];
    const float* W_hn   = (const float*)d_in[10];
    const float* b_n    = (const float*)d_in[11];
    const float* W_io   = (const float*)d_in[12];
    const float* W_ho   = (const float*)d_in[13];
    const float* b_o    = (const float*)d_in[14];
    float* out = (float*)d_out;

    _Float16* hws  = (_Float16*)d_ws;
    _Float16* hact = hws;                       // [2][4096][1024]
    _Float16* hwt  = hws + 2 * (size_t)ACT_SZ;  // [8][1024][1024], n-major

    // pass 1a: activations -> f16
    cvt_act<<<dim3(ACT_SZ / 1024), dim3(256), 0, stream>>>(input, hact);
    cvt_act<<<dim3(ACT_SZ / 1024), dim3(256), 0, stream>>>(hidden, hact + ACT_SZ);

    // pass 1b: weights -> f16, transposed to [n][k]; order f,i,n,o per phase
    cvt_wt<<<dim3(HID / 64, HID / 64, 8), dim3(256), 0, stream>>>(
        W_if, W_ii, W_in, W_io, W_hf, W_hi, W_hn, W_ho, hwt);

    // pass 2: fused LSTM GEMM + epilogue
    lstm_cell_wmma<<<dim3(HID / BN, BATCH_SZ / BM), dim3(256), 0, stream>>>(
        hact, hwt, cell, b_f, b_i, b_n, b_o, out);
}
